// DebugGraphConvEncoder_13417477833482
// MI455X (gfx1250) — matrix operand order repaired and checked by reading, not compiled, not run on hardware
//
#include <hip/hip_runtime.h>
#include <hip/hip_bf16.h>

#define HD 128           // hidden size
#define GG 512           // graphs
#define NPER0 256
#define NE 1048576       // edges
#define N0 (GG * NPER0)  // 131072 nodes

typedef __attribute__((ext_vector_type(16))) _Float16 v16h;
typedef __attribute__((ext_vector_type(8)))  float    v8f;

// ---------------------------------------------------------------------------
// generic fill
__global__ void fill_f32(float* __restrict__ p, float v, int n) {
    int i = blockIdx.x * 256 + threadIdx.x;
    if (i < n) p[i] = v;
}

// ---------------------------------------------------------------------------
// convert W (128x128 f32, row-major [k][n]) -> f16 transposed [n][k]
__global__ void convert_w(const float* __restrict__ W, _Float16* __restrict__ Wt) {
    int i = blockIdx.x * 256 + threadIdx.x;   // 0..16383
    int k = i >> 7, n = i & 127;
    Wt[n * HD + k] = (_Float16)W[k * HD + n];
}

// ---------------------------------------------------------------------------
// Y[nrows x 128] = X[nrows x 128] @ W[128 x 128], f16 WMMA, f32 accumulate.
// Block: 256 threads = 8 waves. Block owns one 16-row tile; wave w owns
// column tile n0 = 16*w. K-loop: 4 x v_wmma_f32_16x16x32_f16.
__global__ __launch_bounds__(256)
void gemm_wmma(const float* __restrict__ X, const _Float16* __restrict__ Wt16,
               float* __restrict__ Y, int nrows) {
    __shared__ _Float16 sW[HD * HD];   // transposed: sW[n*128 + k]
    __shared__ _Float16 sX[16 * HD];   // 16-row tile, row-major

    int tid = threadIdx.x;
    // stage W (32KB) into LDS as dwords: 8192 dwords / 256 threads
    const unsigned int* wsrc = (const unsigned int*)Wt16;
    unsigned int*       wdst = (unsigned int*)sW;
    for (int i = tid; i < HD * HD / 2; i += 256) wdst[i] = wsrc[i];

    int m0 = blockIdx.x * 16;
    for (int i = tid; i < 16 * HD; i += 256) {
        int r = i >> 7, c = i & 127;
        sX[i] = (_Float16)X[(size_t)(m0 + r) * HD + c];
    }
    __syncthreads();

    int wave = tid >> 5;
    int lane = tid & 31;
    int n0   = wave * 16;
    int Mloc = lane & 15;
    int half = lane >> 4;
    int Ncol = n0 + Mloc;

    v8f acc = {};
    #pragma unroll
    for (int k0 = 0; k0 < HD; k0 += 32) {
        v16h a, b;
        // A fragment: element e -> K = k0 + half*16 + e
        const _Float16* ap = &sX[Mloc * HD + k0 + half * 16];
        #pragma unroll
        for (int e = 0; e < 16; ++e) a[e] = ap[e];
        // B fragment: element e -> K = k0 + half*16 + e, column Ncol
        const _Float16* bp = &sW[Ncol * HD + k0 + half * 16];
        #pragma unroll
        for (int e = 0; e < 16; ++e) b[e] = bp[e];
        acc = __builtin_amdgcn_wmma_f32_16x16x32_f16(
            false, a, false, b, (short)0, acc, false, false);
    }
    // D: VGPR r -> row m0 + half*8 + r, col Ncol
    #pragma unroll
    for (int r = 0; r < 8; ++r)
        Y[(size_t)(m0 + half * 8 + r) * HD + Ncol] = acc[r];
}

// ---------------------------------------------------------------------------
// deg[dst] += w_e  (deg pre-filled with 1.0 for the self loop)
__global__ void deg_edges(const int* __restrict__ dst, const float* __restrict__ we,
                          float* __restrict__ deg, int nE) {
    int e = blockIdx.x * 256 + threadIdx.x;
    if (e < nE) atomicAdd(&deg[dst[e]], we[e]);
}

// deg -> rsqrt(deg), in place
__global__ void rsqrt_inplace(float* __restrict__ d, int n) {
    int i = blockIdx.x * 256 + threadIdx.x;
    if (i < n) d[i] = rsqrtf(d[i]);
}

// one wave per edge: agg[dst] += h[src] * (w * dinv[src] * dinv[dst])
__global__ void agg_edges(const int* __restrict__ src, const int* __restrict__ dst,
                          const float* __restrict__ we, const float* __restrict__ dinv,
                          const float* __restrict__ h, float* __restrict__ agg, int nE) {
    int t = blockIdx.x * 256 + threadIdx.x;
    int e = t >> 5;
    if (e >= nE) return;
    float w = we[e];
    if (w == 0.0f) return;
    int lane = t & 31;
    int s = src[e], d = dst[e];
    float norm = w * dinv[s] * dinv[d];
    float4 v = ((const float4*)(h + (size_t)s * HD))[lane];
    float* arow = agg + (size_t)d * HD + lane * 4;
    atomicAdd(arow + 0, v.x * norm);
    atomicAdd(arow + 1, v.y * norm);
    atomicAdd(arow + 2, v.z * norm);
    atomicAdd(arow + 3, v.w * norm);
}

// act = relu(agg + h * dinv^2 + b), written in-place into agg
__global__ void act_kernel(float* __restrict__ agg, const float* __restrict__ h,
                           const float* __restrict__ dinv, const float* __restrict__ b,
                           int total) {
    int t = blockIdx.x * 256 + threadIdx.x;
    if (t >= total) return;
    int node = t >> 7, c = t & 127;
    float di = dinv[node];
    float v = agg[t] + h[t] * di * di + b[c];
    agg[t] = fmaxf(v, 0.0f);
}

// ---------------------------------------------------------------------------
// pinv = 1 / ||p||   (one block, 128 threads)
__global__ void pnorm_kernel(const float* __restrict__ p, float* __restrict__ pinv) {
    __shared__ float red[128];
    int tid = threadIdx.x;
    float v = p[tid];
    red[tid] = v * v;
    __syncthreads();
    for (int s = 64; s > 0; s >>= 1) {
        if (tid < s) red[tid] += red[tid + s];
        __syncthreads();
    }
    if (tid == 0) *pinv = rsqrtf(red[0]);
}

// score[i] = (x[i] . p) * pinv
__global__ void score_kernel(const float* __restrict__ x, const float* __restrict__ p,
                             const float* __restrict__ pinv, float* __restrict__ score, int n) {
    int i = blockIdx.x * 256 + threadIdx.x;
    if (i >= n) return;
    const float4* row = (const float4*)(x + (size_t)i * HD);
    const float4* pp  = (const float4*)p;
    float acc = 0.0f;
    #pragma unroll 8
    for (int j = 0; j < 32; ++j) {
        float4 a = row[j], b = pp[j];
        acc += a.x * b.x + a.y * b.y + a.z * b.z + a.w * b.w;
    }
    score[i] = acc * (*pinv);
}

// one block per graph: rank-select top-k, scale by tanh(score), build nmap
__global__ __launch_bounds__(256)
void topk_select(const float* __restrict__ x, const float* __restrict__ score,
                 int n_per, int k, float* __restrict__ xnew, int* __restrict__ nmap) {
    __shared__ float s[256];
    int g = blockIdx.x, tid = threadIdx.x;
    size_t base = (size_t)g * n_per;
    s[tid] = (tid < n_per) ? score[base + tid] : -3.0e38f;
    __syncthreads();
    if (tid >= n_per) return;
    float mys = s[tid];
    int rank = 0;
    for (int j = 0; j < n_per; ++j) {
        float o = s[j];
        rank += (o > mys) || (o == mys && j < tid);
    }
    if (rank < k) {
        int ni = g * k + rank;
        float scale = tanhf(mys);
        const float4* xi = (const float4*)(x + (base + tid) * (size_t)HD);
        float4*       xo = (float4*)(xnew + (size_t)ni * HD);
        #pragma unroll 4
        for (int c = 0; c < 32; ++c) {
            float4 v = xi[c];
            v.x *= scale; v.y *= scale; v.z *= scale; v.w *= scale;
            xo[c] = v;
        }
        nmap[base + tid] = ni;
    } else {
        nmap[base + tid] = -1;
    }
}

// remap edges through nmap; invalid edges -> idx 0, weight 0
__global__ void remap_edges(const int* __restrict__ srcO, const int* __restrict__ dstO,
                            int* __restrict__ srcN, int* __restrict__ dstN,
                            float* __restrict__ we, const int* __restrict__ nmap, int nE) {
    int e = blockIdx.x * 256 + threadIdx.x;
    if (e >= nE) return;
    int ms = nmap[srcO[e]];
    int md = nmap[dstO[e]];
    bool valid = (ms >= 0) && (md >= 0);
    float w = we[e];
    srcN[e] = valid ? ms : 0;
    dstN[e] = valid ? md : 0;
    we[e]   = valid ? w : 0.0f;
}

// global attention pool, one block (256 threads) per graph
__global__ __launch_bounds__(256)
void attpool(const float* __restrict__ x, int n_per, const float* __restrict__ Wg,
             const float* __restrict__ bgp, float* __restrict__ out, int accumulate) {
    __shared__ float a[256];
    __shared__ float red[256];
    int g = blockIdx.x, tid = threadIdx.x;
    size_t base = (size_t)g * n_per;

    float gate = -3.0e38f;
    if (tid < n_per) {
        const float4* row = (const float4*)(x + (base + tid) * (size_t)HD);
        const float4* wp  = (const float4*)Wg;
        float acc = 0.0f;
        #pragma unroll 8
        for (int j = 0; j < 32; ++j) {
            float4 aa = row[j], bb = wp[j];
            acc += aa.x * bb.x + aa.y * bb.y + aa.z * bb.z + aa.w * bb.w;
        }
        gate = acc + *bgp;
    }
    red[tid] = gate;
    __syncthreads();
    for (int s = 128; s > 0; s >>= 1) {
        if (tid < s) red[tid] = fmaxf(red[tid], red[tid + s]);
        __syncthreads();
    }
    float mx = red[0];
    __syncthreads();
    float ev = (tid < n_per) ? __expf(gate - mx) : 0.0f;
    a[tid]   = ev;
    red[tid] = ev;
    __syncthreads();
    for (int s = 128; s > 0; s >>= 1) {
        if (tid < s) red[tid] += red[tid + s];
        __syncthreads();
    }
    float inv = 1.0f / red[0];
    __syncthreads();
    if (tid < HD) {
        float acc = 0.0f;
        for (int n = 0; n < n_per; ++n)
            acc += a[n] * x[(base + n) * (size_t)HD + tid];
        acc *= inv;
        size_t o = (size_t)g * HD + tid;
        out[o] = accumulate ? out[o] + acc : acc;
    }
}

// ---------------------------------------------------------------------------
extern "C" void kernel_launch(void* const* d_in, const int* in_sizes, int n_in,
                              void* d_out, int out_size, void* d_ws, size_t ws_size,
                              hipStream_t stream) {
    const float* d_x  = (const float*)d_in[0];
    const int*   d_ei = (const int*)d_in[1];
    const float* Wl[3] = { (const float*)d_in[2], (const float*)d_in[5], (const float*)d_in[8] };
    const float* bl[3] = { (const float*)d_in[3], (const float*)d_in[6], (const float*)d_in[9] };
    const float* pl[3] = { (const float*)d_in[4], (const float*)d_in[7], (const float*)d_in[10] };
    const float* Wg  = (const float*)d_in[11];
    const float* bgp = (const float*)d_in[12];
    float* out = (float*)d_out;

    // workspace partition
    char* ws = (char*)d_ws;
    size_t off = 0;
    auto take = [&](size_t bytes) { char* p = ws + off; off += (bytes + 255) & ~(size_t)255; return p; };
    float*     bufH  = (float*)take((size_t)N0 * HD * 4);           // 64 MB  : h = x@W
    float*     bufG  = (float*)take((size_t)N0 * HD * 4);           // 64 MB  : agg / act
    float*     bufP  = (float*)take((size_t)GG * 205 * HD * 4);     // 54 MB  : pooled feats
    int*       wsSrc = (int*)take((size_t)NE * 4);
    int*       wsDst = (int*)take((size_t)NE * 4);
    float*     wsWe  = (float*)take((size_t)NE * 4);
    float*     deg   = (float*)take((size_t)N0 * 4);                // deg -> dinv in place
    float*     score = (float*)take((size_t)N0 * 4);
    int*       nmap  = (int*)take((size_t)N0 * 4);
    _Float16*  w16   = (_Float16*)take((size_t)HD * HD * 2);
    float*     pinv  = (float*)take(256);
    (void)ws_size; (void)in_sizes; (void)n_in; (void)out_size;

    const int nper_in[3] = { 256, 205, 164 };
    const int k_out[3]   = { 205, 164, 132 };

    // edge weights start as ones
    fill_f32<<<NE / 256, 256, 0, stream>>>(wsWe, 1.0f, NE);

    for (int l = 0; l < 3; ++l) {
        int nper   = nper_in[l];
        int nnodes = GG * nper;
        int k      = k_out[l];
        const float* xcur = (l == 0) ? d_x : bufP;
        const int* srcOld = (l == 0) ? d_ei        : wsSrc;
        const int* dstOld = (l == 0) ? (d_ei + NE) : wsDst;

        // h = xcur @ W_l  (WMMA)
        convert_w<<<64, 256, 0, stream>>>(Wl[l], w16);
        gemm_wmma<<<nnodes / 16, 256, 0, stream>>>(xcur, w16, bufH, nnodes);

        // degree (self loop = 1) and dinv
        fill_f32<<<(nnodes + 255) / 256, 256, 0, stream>>>(deg, 1.0f, nnodes);
        deg_edges<<<NE / 256, 256, 0, stream>>>(dstOld, wsWe, deg, NE);
        rsqrt_inplace<<<(nnodes + 255) / 256, 256, 0, stream>>>(deg, nnodes);

        // normalized aggregation + self loop + bias + relu
        fill_f32<<<(nnodes * HD) / 256, 256, 0, stream>>>(bufG, 0.0f, nnodes * HD);
        agg_edges<<<(NE * 32) / 256, 256, 0, stream>>>(srcOld, dstOld, wsWe, deg, bufH, bufG, NE);
        act_kernel<<<(nnodes * HD) / 256, 256, 0, stream>>>(bufG, bufH, deg, bl[l], nnodes * HD);

        // TopK pooling
        pnorm_kernel<<<1, 128, 0, stream>>>(pl[l], pinv);
        score_kernel<<<(nnodes + 255) / 256, 256, 0, stream>>>(bufG, pl[l], pinv, score, nnodes);
        topk_select<<<GG, 256, 0, stream>>>(bufG, score, nper, k, bufP, nmap);
        if (l < 2)
            remap_edges<<<NE / 256, 256, 0, stream>>>(srcOld, dstOld, wsSrc, wsDst, wsWe, nmap, NE);

        // attention pool, accumulate into output
        attpool<<<GG, 256, 0, stream>>>(bufP, k, Wg, bgp, out, l > 0 ? 1 : 0);
    }
}
